// SpikingSelfAttention_86225763434763
// MI455X (gfx1250) — 16-bit and 8-bit matrix operand order repaired and checked by reading, compiled, not run on hardware
//
#include <hip/hip_runtime.h>
#include <hip/hip_fp16.h>

typedef __attribute__((ext_vector_type(16))) _Float16 v16h;
typedef __attribute__((ext_vector_type(8)))  _Float16 v8h;
typedef __attribute__((ext_vector_type(8)))  float    v8f;
typedef __attribute__((ext_vector_type(8)))  int      v8i;
typedef __attribute__((ext_vector_type(4)))  float    f4;
typedef __attribute__((ext_vector_type(4)))  unsigned int u32x4;
typedef __attribute__((ext_vector_type(8)))  int      i32x8;
typedef __attribute__((ext_vector_type(4)))  int      i32x4;

#define T_  4
#define B_  32
#define C_  384
#define N_  256
#define H_  8
#define D_  48
#define DP_ 64
#define TB_ (T_*B_)

__device__ __forceinline__ void tdm_copy_1d(const void* gptr, unsigned lds_off,
                                            unsigned bytes) {
  const unsigned long long ga = (unsigned long long)gptr;
  const unsigned n8 = bytes >> 3;
  const u32x4 g0 = { 1u,
                     lds_off,
                     (unsigned)(ga & 0xffffffffu),
                     (unsigned)((ga >> 32) & 0x01ffffffu) | (2u << 30) };
  const i32x8 g1 = { (int)(3u << 16),
                     (int)((n8 & 0xffffu) << 16),
                     (int)(((n8 >> 16) & 0xffffu) | (1u << 16)),
                     (int)((n8 & 0xffffu) << 16),
                     0,
                     (int)n8,
                     0, 0 };
  const i32x4 z4 = {0, 0, 0, 0};
#if __clang_major__ >= 23
  const i32x8 z8 = {0, 0, 0, 0, 0, 0, 0, 0};
  __builtin_amdgcn_tensor_load_to_lds(g0, g1, z4, z4, z8, 0);
#else
  __builtin_amdgcn_tensor_load_to_lds(g0, g1, z4, z4, 0);
#endif
}

__device__ __forceinline__ unsigned lds_off_of(const void* p) {
  return (unsigned)(unsigned long long)p;
}

__global__ __launch_bounds__(256) void lif_transpose_f16(
    const float* __restrict__ x, _Float16* __restrict__ s)
{
  __shared__ _Float16 tile[T_][32][33];
  const int n0 = blockIdx.x * 32, c0 = blockIdx.y * 32, b = blockIdx.z;
  const int nl = threadIdx.x & 31, cs = threadIdx.x >> 5;
#pragma unroll
  for (int ci = 0; ci < 4; ++ci) {
    const int cl = cs * 4 + ci;
    float v = 0.f;
#pragma unroll
    for (int t = 0; t < T_; ++t) {
      const float xv = x[(((size_t)t * B_ + b) * C_ + c0 + cl) * N_ + n0 + nl];
      const float h  = 0.5f * (v + xv);
      const float sp = (h >= 1.0f) ? 1.f : 0.f;
      v = h * (1.f - sp);
      tile[t][cl][nl] = (_Float16)sp;
    }
  }
  __syncthreads();
  const int cl2 = threadIdx.x & 31, ns = threadIdx.x >> 5;
#pragma unroll
  for (int ni = 0; ni < 4; ++ni) {
    const int nl2 = ns * 4 + ni;
#pragma unroll
    for (int t = 0; t < T_; ++t)
      s[(((size_t)t * B_ + b) * N_ + n0 + nl2) * C_ + c0 + cl2] = tile[t][cl2][nl2];
  }
}

__global__ void f32_to_f16(const float* __restrict__ src,
                           _Float16* __restrict__ dst, int n) {
  const int i = blockIdx.x * blockDim.x + threadIdx.x;
  if (i < n) dst[i] = (_Float16)src[i];
}

__global__ __launch_bounds__(256) void gemm_bn_f16(
    const _Float16* __restrict__ Wf, const _Float16* __restrict__ S,
    const float* __restrict__ gamma, const float* __restrict__ beta,
    const float* __restrict__ mean,  const float* __restrict__ var,
    const float* __restrict__ bias,  float* __restrict__ out)
{
  __shared__ __align__(64) _Float16 sB[64 * C_];
  const int tb   = blockIdx.z;
  const int n0   = blockIdx.x * 64;
  const int m0   = blockIdx.y * 128 + (threadIdx.x >> 5) * 16;
  const int lane = threadIdx.x & 31;
  const int lh   = (lane < 16) ? 0 : 1;

  if (threadIdx.x < 32) {
    tdm_copy_1d(S + ((size_t)tb * N_ + n0) * C_, lds_off_of(sB), 64 * C_ * 2);
    __builtin_amdgcn_s_wait_tensorcnt(0);
  }
  __syncthreads();

  const _Float16* wrow = Wf + (size_t)(m0 + (lane & 15)) * C_;
  __builtin_prefetch(wrow, 0, 1);
  v8f acc[4] = {};
  for (int k0 = 0; k0 < C_; k0 += 32) {
    const v16h a = *(const v16h*)(wrow + k0 + lh * 16);
#pragma unroll
    for (int ct = 0; ct < 4; ++ct) {
      const int coll = ct * 16 + (lane & 15);
      const v16h b = *(const v16h*)(sB + (size_t)coll * C_ + k0 + lh * 16);
      acc[ct] = __builtin_amdgcn_wmma_f32_16x16x32_f16(
          false, a, false, b, (short)0, acc[ct], false, false);
    }
  }
  float invv[8], shf[8];
#pragma unroll
  for (int e = 0; e < 8; ++e) {
    const int row = m0 + e + lh * 8;
    const float iv = gamma[row] * rsqrtf(var[row] + 1e-5f);
    invv[e] = iv;
    shf[e]  = beta[row] - mean[row] * iv + (bias ? bias[row] * iv : 0.f);
  }
#pragma unroll
  for (int ct = 0; ct < 4; ++ct) {
    const int col = n0 + ct * 16 + (lane & 15);
#pragma unroll
    for (int e = 0; e < 8; ++e) {
      const int row = m0 + e + lh * 8;
      out[((size_t)tb * C_ + row) * N_ + col] = acc[ct][e] * invv[e] + shf[e];
    }
  }
}

__global__ __launch_bounds__(256) void lif_qk_u8(
    const float* __restrict__ pre, unsigned char* __restrict__ spk)
{
  __shared__ unsigned char tile[T_][D_][33];
  const int n0 = blockIdx.x * 32, h = blockIdx.y, b = blockIdx.z;
  const int nl = threadIdx.x & 31, js = threadIdx.x >> 5;
#pragma unroll
  for (int ji = 0; ji < 6; ++ji) {
    const int j = js * 6 + ji;
    float v = 0.f;
#pragma unroll
    for (int t = 0; t < T_; ++t) {
      const float xv = pre[(((size_t)t * B_ + b) * C_ + h * D_ + j) * N_ + n0 + nl];
      const float hh = 0.5f * (v + xv);
      const float sp = (hh >= 1.f) ? 1.f : 0.f;
      v = hh * (1.f - sp);
      tile[t][j][nl] = (unsigned char)sp;
    }
  }
  __syncthreads();
  const int j2 = threadIdx.x & 63, ns = threadIdx.x >> 6;
#pragma unroll
  for (int ni = 0; ni < 8; ++ni) {
    const int nl2 = ns * 8 + ni;
#pragma unroll
    for (int t = 0; t < T_; ++t) {
      const unsigned char val = (j2 < D_) ? tile[t][j2][nl2] : (unsigned char)0;
      spk[((((size_t)t * B_ + b) * H_ + h) * N_ + n0 + nl2) * DP_ + j2] = val;
    }
  }
}

__global__ __launch_bounds__(256) void lif_v_u8(
    const float* __restrict__ pre, unsigned char* __restrict__ spk)
{
  const size_t gid = (size_t)blockIdx.x * 256 + threadIdx.x;
  const int n = (int)(gid & (N_ - 1));
  const int c = (int)((gid >> 8) % C_);
  const int b = (int)(gid / ((size_t)N_ * C_));
  const int h = c / D_, j = c % D_;
  float v = 0.f;
#pragma unroll
  for (int t = 0; t < T_; ++t) {
    const float xv = pre[(((size_t)t * B_ + b) * C_ + c) * N_ + n];
    const float hh = 0.5f * (v + xv);
    const float sp = (hh >= 1.f) ? 1.f : 0.f;
    v = hh * (1.f - sp);
    spk[((((size_t)t * B_ + b) * H_ + h) * D_ + j) * N_ + n] = (unsigned char)sp;
  }
}

__global__ __launch_bounds__(256) void attn_fused_iu8(
    const unsigned char* __restrict__ q, const unsigned char* __restrict__ k,
    const unsigned char* __restrict__ vv, float* __restrict__ out)
{
  __shared__ __align__(16) unsigned char attn_s[128 * N_];
  __shared__ __align__(16) unsigned char k_s[N_ * DP_];
  __shared__ __align__(16) unsigned char v_s[D_ * N_];
  const int tbh   = blockIdx.y;
  const int rbase = blockIdx.x * 128;
  const int w = threadIdx.x >> 5, lane = threadIdx.x & 31;
  const int so = (lane < 16) ? 0 : 2;
  const unsigned char* qb = q + (size_t)tbh * N_ * DP_;

  if (threadIdx.x < 32) {
    tdm_copy_1d(k + (size_t)tbh * N_ * DP_, lds_off_of(k_s), N_ * DP_);
    tdm_copy_1d(vv + (size_t)tbh * D_ * N_, lds_off_of(v_s), D_ * N_);
    __builtin_amdgcn_s_wait_tensorcnt(0);
  }
  __syncthreads();

  {
    const int arow = rbase + w * 16 + (lane & 15);
    const int* qp = (const int*)(qb + (size_t)arow * DP_);
    const v8i a = { qp[0+2*so], qp[1+2*so], qp[2+2*so], qp[3+2*so],
                    qp[8+2*so], qp[9+2*so], qp[10+2*so], qp[11+2*so] };
#pragma unroll
    for (int ct = 0; ct < 16; ++ct) {
      const int mcol = ct * 16 + (lane & 15);
      const int* kp = (const int*)(k_s + (size_t)mcol * DP_ + ((lane < 16) ? 0 : 16));
      const v8i bfr = { kp[0], kp[1], kp[2], kp[3], kp[8], kp[9], kp[10], kp[11] };
      v8i acc = {};
      acc = __builtin_amdgcn_wmma_i32_16x16x64_iu8(false, a, false, bfr, acc, false, false);
#pragma unroll
      for (int e = 0; e < 8; ++e) {
        const int rl = w * 16 + e + ((lane < 16) ? 0 : 8);
        attn_s[rl * N_ + mcol] = (unsigned char)acc[e];
      }
    }
  }
  __syncthreads();

  const int tb = tbh >> 3, h = tbh & 7;
#pragma unroll
  for (int jt = 0; jt < 3; ++jt) {
    v8i acc = {};
    const int j    = jt * 16 + (lane & 15);
    const int arow = w * 16 + (lane & 15);
#pragma unroll
    for (int km = 0; km < 4; ++km) {
      const int* lp = (const int*)(attn_s + arow * N_ + km * 64);
      const v8i a = { lp[0+2*so], lp[1+2*so], lp[2+2*so], lp[3+2*so],
                      lp[8+2*so], lp[9+2*so], lp[10+2*so], lp[11+2*so] };
      const int* vp = (const int*)(v_s + (size_t)j * N_ + km * 64 + ((lane < 16) ? 0 : 16));
      const v8i bfr = { vp[0], vp[1], vp[2], vp[3], vp[8], vp[9], vp[10], vp[11] };
      acc = __builtin_amdgcn_wmma_i32_16x16x64_iu8(false, a, false, bfr, acc, false, false);
    }
    const int nb = rbase + w * 16 + ((lane < 16) ? 0 : 8);
    float* op = out + ((size_t)tb * C_ + h * D_ + j) * N_ + nb;
    const f4 o0 = { acc[0]*0.125f, acc[1]*0.125f, acc[2]*0.125f, acc[3]*0.125f };
    const f4 o1 = { acc[4]*0.125f, acc[5]*0.125f, acc[6]*0.125f, acc[7]*0.125f };
    *(f4*)(op)     = o0;
    *(f4*)(op + 4) = o1;
  }
}

extern "C" void kernel_launch(void* const* d_in, const int* in_sizes, int n_in,
                              void* d_out, int out_size, void* d_ws, size_t ws_size,
                              hipStream_t stream) {
  (void)in_sizes; (void)n_in; (void)out_size; (void)ws_size;
  const float* x      = (const float*)d_in[0];
  const float* q_w    = (const float*)d_in[2];
  const float* q_bn[4]= {(const float*)d_in[3],(const float*)d_in[4],(const float*)d_in[5],(const float*)d_in[6]};
  const float* k_w    = (const float*)d_in[7];
  const float* k_bn[4]= {(const float*)d_in[8],(const float*)d_in[9],(const float*)d_in[10],(const float*)d_in[11]};
  const float* v_w    = (const float*)d_in[12];
  const float* v_bn[4]= {(const float*)d_in[13],(const float*)d_in[14],(const float*)d_in[15],(const float*)d_in[16]};
  const float* p_w    = (const float*)d_in[17];
  const float* p_b    = (const float*)d_in[18];
  const float* p_bn[4]= {(const float*)d_in[19],(const float*)d_in[20],(const float*)d_in[21],(const float*)d_in[22]};

  char* ws = (char*)d_ws;
  const size_t S0B  = (size_t)TB_ * N_ * C_ * 2;
  const size_t WB   = (size_t)C_ * C_ * 2;
  const size_t PREB = (size_t)T_ * B_ * C_ * N_ * 4;
  const size_t QKB  = (size_t)TB_ * H_ * N_ * DP_;
  const size_t VSB  = (size_t)TB_ * H_ * D_ * N_;
  size_t o = 0;
  _Float16* s0   = (_Float16*)(ws + o); o += S0B;
  _Float16* Wq   = (_Float16*)(ws + o); o += WB;
  _Float16* Wk   = (_Float16*)(ws + o); o += WB;
  _Float16* Wv   = (_Float16*)(ws + o); o += WB;
  _Float16* Wp   = (_Float16*)(ws + o); o += WB;
  float* q_pre   = (float*)(ws + o); o += PREB;
  float* k_pre   = (float*)(ws + o); o += PREB;
  float* v_pre   = (float*)(ws + o); o += PREB;
  unsigned char* q_spk = (unsigned char*)(ws + o); o += QKB;
  unsigned char* k_spk = (unsigned char*)(ws + o); o += QKB;
  unsigned char* v_spk = (unsigned char*)(ws + o); o += VSB;
  float*    attn_out = q_pre;
  _Float16* s1       = (_Float16*)k_pre;

  const int WN = C_ * C_;
  f32_to_f16<<<(WN + 255) / 256, 256, 0, stream>>>(q_w, Wq, WN);
  f32_to_f16<<<(WN + 255) / 256, 256, 0, stream>>>(k_w, Wk, WN);
  f32_to_f16<<<(WN + 255) / 256, 256, 0, stream>>>(v_w, Wv, WN);
  f32_to_f16<<<(WN + 255) / 256, 256, 0, stream>>>(p_w, Wp, WN);

  const dim3 lifg(N_ / 32, C_ / 32, B_);
  lif_transpose_f16<<<lifg, 256, 0, stream>>>(x, s0);

  const dim3 gg(N_ / 64, C_ / 128, TB_);
  gemm_bn_f16<<<gg, 256, 0, stream>>>(Wq, s0, q_bn[0], q_bn[1], q_bn[2], q_bn[3], nullptr, q_pre);
  gemm_bn_f16<<<gg, 256, 0, stream>>>(Wk, s0, k_bn[0], k_bn[1], k_bn[2], k_bn[3], nullptr, k_pre);
  gemm_bn_f16<<<gg, 256, 0, stream>>>(Wv, s0, v_bn[0], v_bn[1], v_bn[2], v_bn[3], nullptr, v_pre);

  const dim3 qkg(N_ / 32, H_, B_);
  lif_qk_u8<<<qkg, 256, 0, stream>>>(q_pre, q_spk);
  lif_qk_u8<<<qkg, 256, 0, stream>>>(k_pre, k_spk);
  lif_v_u8<<<(B_ * C_ * N_) / 256, 256, 0, stream>>>(v_pre, v_spk);

  attn_fused_iu8<<<dim3(2, T_ * B_ * H_), 256, 0, stream>>>(q_spk, k_spk, v_spk, attn_out);

  lif_transpose_f16<<<lifg, 256, 0, stream>>>(attn_out, s1);
  gemm_bn_f16<<<gg, 256, 0, stream>>>(Wp, s1, p_bn[0], p_bn[1], p_bn[2], p_bn[3], p_b, (float*)d_out);
}
